// Attention_38577396253121
// MI455X (gfx1250) — matrix operand order repaired and checked by reading, not compiled, not run on hardware
//
#include <hip/hip_runtime.h>
#include <hip/hip_bf16.h>

#define NX 768
#define NHEAD 12
#define DH 64
#define SEQ 2048
#define BATCH 2
#define N3 (3*NX)          // 2304
#define MTOT (BATCH*SEQ)   // 4096
#define LORA_SCALE 16.0f

#define SA 40              // LDS row stride (shorts) for 32-wide tiles (80B, 16B-aligned)
#define SK 72              // LDS row stride (shorts) for 64-wide K tile (144B, 16B-aligned)

typedef __bf16 bf16_t;
typedef bf16_t v16bf __attribute__((ext_vector_type(16)));
typedef float  v8f   __attribute__((ext_vector_type(8)));
typedef int    v4i   __attribute__((ext_vector_type(4)));
typedef unsigned int uint4v __attribute__((ext_vector_type(4)));
typedef unsigned short ushort_t;

union Frag {
    v16bf v;
    unsigned int   u[8];
    unsigned short s[16];
};

// ---------------------------------------------------------------------------
// CDNA5 async global->LDS staging (guarded; falls back to sync b128 copies)
// ---------------------------------------------------------------------------
#if defined(__gfx1250__) && __has_builtin(__builtin_amdgcn_global_load_async_to_lds_b128) && __has_builtin(__builtin_amdgcn_s_wait_asynccnt)
#define USE_ASYNC 1
#else
#define USE_ASYNC 0
#endif

__device__ __forceinline__ void copy16_to_lds(ushort_t* ldsp, const ushort_t* gp) {
#if USE_ASYNC
    __builtin_amdgcn_global_load_async_to_lds_b128(
        (__attribute__((address_space(1))) v4i*)(uintptr_t)gp,
        (__attribute__((address_space(3))) v4i*)ldsp, 0, 0);
#else
    *(uint4v*)ldsp = *(const uint4v*)gp;
#endif
}

__device__ __forceinline__ void stage_join() {
#if USE_ASYNC
    __builtin_amdgcn_s_wait_asynccnt(0);
#endif
    __syncthreads();
}

// round-to-nearest-even fp32 -> bf16 bits
__device__ __forceinline__ ushort_t f2bf(float x) {
    unsigned int u = __float_as_uint(x);
    return (ushort_t)((u + 0x7FFFu + ((u >> 16) & 1u)) >> 16);
}

// error-compensated split: x ~= hi + lo (both bf16)
__device__ __forceinline__ void splitbf(float x, ushort_t &hi, ushort_t &lo) {
    unsigned int u = __float_as_uint(x);
    unsigned int h = (u + 0x7FFFu + ((u >> 16) & 1u)) >> 16;
    hi = (ushort_t)h;
    float hf = __uint_as_float(h << 16);
    lo = f2bf(x - hf);
}

__device__ __forceinline__ v8f wmma_bf(const Frag &a, const Frag &b, v8f c) {
    return __builtin_amdgcn_wmma_f32_16x16x32_bf16(false, a.v, false, b.v,
                                                   (short)0, c, false, false);
}

// ---------------------------------------------------------------------------
// Prep 1: split x -> bf16 hi/lo (row-major [M][K])
// ---------------------------------------------------------------------------
__global__ void __launch_bounds__(256)
prep_split(const float* __restrict__ src, ushort_t* __restrict__ hi,
           ushort_t* __restrict__ lo, int n) {
    int idx = blockIdx.x * 256 + threadIdx.x;
    if (idx >= n) return;
    ushort_t h, l; splitbf(src[idx], h, l);
    hi[idx] = h; lo[idx] = l;
}

// ---------------------------------------------------------------------------
// Prep 2: W_eff^T = (w_attn + LORA_SCALE*(B_c A_c)^T)^T, stored [n][k] bf16 hi/lo
// ---------------------------------------------------------------------------
__global__ void __launch_bounds__(256)
weffT_kernel(const float* __restrict__ w_attn, const float* __restrict__ lora_A,
             const float* __restrict__ lora_B, ushort_t* __restrict__ bthi,
             ushort_t* __restrict__ btlo) {
    int idx = blockIdx.x * 256 + threadIdx.x;
    if (idx >= N3 * NX) return;
    int n = idx / NX;
    int k = idx - n * NX;
    int c = n / NX;
    float s = w_attn[(size_t)k * N3 + n];
#pragma unroll
    for (int r = 0; r < 8; ++r)
        s += LORA_SCALE * lora_B[n * 8 + r] * lora_A[(c * 8 + r) * NX + k];
    ushort_t h, l; splitbf(s, h, l);
    bthi[idx] = h; btlo[idx] = l;
}

// ---------------------------------------------------------------------------
// Prep 3: transpose+split w_proj [K][N] -> [n][k] bf16 hi/lo
// ---------------------------------------------------------------------------
__global__ void __launch_bounds__(256)
wT_kernel(const float* __restrict__ w, ushort_t* __restrict__ hi,
          ushort_t* __restrict__ lo, int K, int N) {
    int idx = blockIdx.x * 256 + threadIdx.x;
    if (idx >= K * N) return;
    int n = idx / K;
    int k = idx - n * K;
    ushort_t h, l; splitbf(w[(size_t)k * N + n], h, l);
    hi[idx] = h; lo[idx] = l;
}

// ---------------------------------------------------------------------------
// Prep 4 (after QKV): transpose V [bh][s][d] fp32 -> vt [bh][d][s] bf16 hi/lo
// ---------------------------------------------------------------------------
__global__ void __launch_bounds__(256)
vtrans_kernel(const float* __restrict__ vp, ushort_t* __restrict__ vthi,
              ushort_t* __restrict__ vtlo) {
    __shared__ float tile[64][65];
    const int bh = blockIdx.y;
    const int s0 = blockIdx.x * 64;
    const int tid = threadIdx.x;
#pragma unroll
    for (int t = 0; t < 16; ++t) {
        int i = t * 256 + tid;
        int s = i >> 6, d = i & 63;
        tile[s][d] = vp[((size_t)bh * SEQ + s0 + s) * DH + d];
    }
    __syncthreads();
#pragma unroll
    for (int t = 0; t < 16; ++t) {
        int i = t * 256 + tid;
        int d = i >> 6, s = i & 63;
        ushort_t h, l; splitbf(tile[s][d], h, l);
        size_t o = ((size_t)bh * DH + d) * SEQ + s0 + s;
        vthi[o] = h; vtlo[o] = l;
    }
}

// ---------------------------------------------------------------------------
// GEMM out = A[MxK] @ B[KxN] + bias. A/B pre-split bf16, B transposed [n][k].
// MODE 0: QKV scatter (q*0.125 fp32, k fp32 + bf16 hi/lo, v fp32)
// MODE 1: plain fp32 row-major store
// ---------------------------------------------------------------------------
template <int MODE>
__global__ void __launch_bounds__(256)
gemm_kernel(const ushort_t* __restrict__ Ahi, const ushort_t* __restrict__ Alo,
            const ushort_t* __restrict__ Bthi, const ushort_t* __restrict__ Btlo,
            const float* __restrict__ bias,
            float* __restrict__ out0, float* __restrict__ out1,
            float* __restrict__ out2,
            ushort_t* __restrict__ khi, ushort_t* __restrict__ klo,
            int N, int K) {
    __shared__ alignas(16) ushort_t a_hi[64 * SA], a_lo[64 * SA];
    __shared__ alignas(16) ushort_t b_hi[64 * SA], b_lo[64 * SA];

    const int tid  = threadIdx.x;
    const int lane = tid & 31;
    const int w    = tid >> 5;
    const int half = lane >> 4;
    const int ln   = lane & 15;
    const int m0   = blockIdx.y * 64;
    const int n0   = blockIdx.x * 64;
    const int mt   = w & 3;
    const int ntb  = (w >> 2) * 2;

    // staging map: thread -> (row 0..63, 16B chunk 0..3)
    const int srow = tid >> 2;
    const int sch  = (tid & 3) * 8;   // shorts

    v8f acc0 = {}; v8f acc1 = {};

    const int KC = K >> 5;
    for (int kc = 0; kc < KC; ++kc) {
        const size_t ga = (size_t)(m0 + srow) * K + kc * 32 + sch;
        const size_t gb = (size_t)(n0 + srow) * K + kc * 32 + sch;
        copy16_to_lds(&a_hi[srow * SA + sch], Ahi + ga);
        copy16_to_lds(&a_lo[srow * SA + sch], Alo + ga);
        copy16_to_lds(&b_hi[srow * SA + sch], Bthi + gb);
        copy16_to_lds(&b_lo[srow * SA + sch], Btlo + gb);
        stage_join();

        Frag ah, al;
        {
            const unsigned int* ph = (const unsigned int*)&a_hi[(mt * 16 + ln) * SA];
            const unsigned int* pl = (const unsigned int*)&a_lo[(mt * 16 + ln) * SA];
#pragma unroll
            for (int j = 0; j < 8; ++j) {
                int off = j + ((j >= 4) ? 4 : 0) + 4 * half;
                ah.u[j] = ph[off]; al.u[j] = pl[off];
            }
        }
#pragma unroll
        for (int nt = 0; nt < 2; ++nt) {
            Frag bh, bl;
            const unsigned int* ph = (const unsigned int*)&b_hi[((ntb + nt) * 16 + ln) * SA];
            const unsigned int* pl = (const unsigned int*)&b_lo[((ntb + nt) * 16 + ln) * SA];
#pragma unroll
            for (int j = 0; j < 8; ++j) {
                int off = j + ((j >= 4) ? 4 : 0) + 4 * half;
                bh.u[j] = ph[off]; bl.u[j] = pl[off];
            }
            v8f &acc = nt ? acc1 : acc0;
            acc = wmma_bf(ah, bh, acc);
            acc = wmma_bf(al, bh, acc);
            acc = wmma_bf(ah, bl, acc);
        }
        __syncthreads();
    }

#pragma unroll
    for (int nt = 0; nt < 2; ++nt) {
        const v8f acc = nt ? acc1 : acc0;
        int n_g = n0 + (ntb + nt) * 16 + ln;
        float bv = bias[n_g];
#pragma unroll
        for (int v = 0; v < 8; ++v) {
            int m_g = m0 + mt * 16 + v + 8 * half;
            float val = acc[v] + bv;
            if (MODE == 0) {
                int c  = n_g / NX;
                int e  = n_g - c * NX;
                int hh = e >> 6, d = e & 63;
                int bb = m_g >> 11, s = m_g & 2047;
                size_t idx = ((size_t)((bb * NHEAD + hh) * SEQ + s)) * DH + d;
                if (c == 0) {
                    out0[idx] = val * 0.125f;          // q, folded 1/sqrt(dh)
                } else if (c == 1) {
                    out1[idx] = val;                   // k -> present[0]
                    ushort_t h, l; splitbf(val, h, l); // k bf16 copy for attention
                    khi[idx] = h; klo[idx] = l;
                } else {
                    out2[idx] = val;                   // v -> present[1]
                }
            } else {
                out0[(size_t)m_g * N + n_g] = val;
            }
        }
    }
}

// ---------------------------------------------------------------------------
// Causal flash attention. 128 queries/block, 32-key blocks, bf16x3 WMMA,
// K/V pre-split bf16 staged via async copies; output written pre-split bf16.
// ---------------------------------------------------------------------------
__global__ void __launch_bounds__(256)
attn_kernel(const float* __restrict__ qws,
            const ushort_t* __restrict__ khi, const ushort_t* __restrict__ klo,
            const ushort_t* __restrict__ vthi, const ushort_t* __restrict__ vtlo,
            ushort_t* __restrict__ ahi, ushort_t* __restrict__ alo) {
    __shared__ alignas(16) ushort_t k_hi[32 * SK], k_lo[32 * SK];
    __shared__ alignas(16) ushort_t vt_hi[64 * SA], vt_lo[64 * SA];
    __shared__ alignas(16) ushort_t p_hi[8 * 16 * SA], p_lo[8 * 16 * SA];

    const int tid  = threadIdx.x;
    const int lane = tid & 31;
    const int w    = tid >> 5;
    const int half = lane >> 4;
    const int ln   = lane & 15;
    const int bh   = blockIdx.y;
    const int q0   = blockIdx.x * 128;
    const size_t base = (size_t)bh * SEQ * DH;

    // loop-invariant Q fragments (fp32 -> bf16 hi/lo, once per block)
    Frag qh[2], ql[2];
    {
        const float* qrow = qws + base + (size_t)(q0 + w * 16 + ln) * DH;
#pragma unroll
        for (int kc = 0; kc < 2; ++kc) {
#pragma unroll
            for (int e = 0; e < 16; ++e) {
                int k = kc * 32 + e + (e & 8) + 8 * half;
                ushort_t h, l; splitbf(qrow[k], h, l);
                qh[kc].s[e] = h; ql[kc].s[e] = l;
            }
        }
    }

    v8f o0 = {}, o1 = {}, o2 = {}, o3 = {};
    float mrun[8], lrun[8];
#pragma unroll
    for (int v = 0; v < 8; ++v) { mrun[v] = -3.0e38f; lrun[v] = 0.f; }

    // staging maps
    const int krow = tid >> 3, kch = (tid & 7) * 8;   // K: 32 rows x 8 chunks
    const int vrow = tid >> 2, vch = (tid & 3) * 8;   // Vt: 64 rows x 4 chunks

    const int jmax = q0 / 32 + 4;
    for (int j = 0; j < jmax; ++j) {
        const size_t gk = base + (size_t)(j * 32 + krow) * DH + kch;
        const size_t gv = ((size_t)bh * DH + vrow) * SEQ + j * 32 + vch;
        copy16_to_lds(&k_hi[krow * SK + kch], khi + gk);
        copy16_to_lds(&k_lo[krow * SK + kch], klo + gk);
        copy16_to_lds(&vt_hi[vrow * SA + vch], vthi + gv);
        copy16_to_lds(&vt_lo[vrow * SA + vch], vtlo + gv);
        stage_join();

        // logits S = (q/8).k over two 16-key tiles
        float S0[8], S1[8];
#pragma unroll
        for (int nt = 0; nt < 2; ++nt) {
            v8f sa = {};
#pragma unroll
            for (int kc = 0; kc < 2; ++kc) {
                Frag kb, kl2;
                const unsigned int* ph = (const unsigned int*)&k_hi[(nt * 16 + ln) * SK];
                const unsigned int* pl = (const unsigned int*)&k_lo[(nt * 16 + ln) * SK];
#pragma unroll
                for (int jj = 0; jj < 8; ++jj) {
                    int off = kc * 16 + jj + ((jj >= 4) ? 4 : 0) + 4 * half;
                    kb.u[jj] = ph[off]; kl2.u[jj] = pl[off];
                }
                sa = wmma_bf(qh[kc], kb, sa);
                sa = wmma_bf(ql[kc], kb, sa);
                sa = wmma_bf(qh[kc], kl2, sa);
            }
            float* Sd = nt ? S1 : S0;
#pragma unroll
            for (int v = 0; v < 8; ++v) Sd[v] = sa[v];
        }

        // causal mask
        const int qgb = q0 + w * 16;
#pragma unroll
        for (int v = 0; v < 8; ++v) {
            int qg = qgb + v + 8 * half;
            int kg = j * 32 + ln;
            if (kg > qg)      S0[v] = -3.0e38f;
            if (kg + 16 > qg) S1[v] = -3.0e38f;
        }

        // online softmax
        float alpha[8];
#pragma unroll
        for (int v = 0; v < 8; ++v) {
            float m2 = fmaxf(S0[v], S1[v]);
#pragma unroll
            for (int sh = 8; sh >= 1; sh >>= 1)
                m2 = fmaxf(m2, __shfl_xor(m2, sh, 32));
            float mn = fmaxf(mrun[v], m2);
            alpha[v] = __expf(mrun[v] - mn);
            mrun[v]  = mn;
            lrun[v] *= alpha[v];
        }

#pragma unroll
        for (int v = 0; v < 8; ++v) {
            float p0 = __expf(S0[v] - mrun[v]);
            float p1 = __expf(S1[v] - mrun[v]);
            ushort_t h, l;
            int prow = (w * 16 + v + 8 * half) * SA;
            splitbf(p0, h, l);
            p_hi[prow + ln] = h;      p_lo[prow + ln] = l;
            splitbf(p1, h, l);
            p_hi[prow + 16 + ln] = h; p_lo[prow + 16 + ln] = l;
            float rs = p0 + p1;
#pragma unroll
            for (int sh = 8; sh >= 1; sh >>= 1)
                rs += __shfl_xor(rs, sh, 32);
            lrun[v] += rs;
        }

#pragma unroll
        for (int v = 0; v < 8; ++v) {
            o0[v] *= alpha[v]; o1[v] *= alpha[v];
            o2[v] *= alpha[v]; o3[v] *= alpha[v];
        }

        // P fragments (A-layout) from wave-private LDS
        Frag pa, pl2;
        {
            const unsigned int* ph = (const unsigned int*)&p_hi[(w * 16 + ln) * SA];
            const unsigned int* pl = (const unsigned int*)&p_lo[(w * 16 + ln) * SA];
#pragma unroll
            for (int jj = 0; jj < 8; ++jj) {
                int off = jj + ((jj >= 4) ? 4 : 0) + 4 * half;
                pa.u[jj] = ph[off]; pl2.u[jj] = pl[off];
            }
        }
        // O += P @ V
#pragma unroll
        for (int dt = 0; dt < 4; ++dt) {
            Frag vb, vl;
            const unsigned int* ph = (const unsigned int*)&vt_hi[(dt * 16 + ln) * SA];
            const unsigned int* pl = (const unsigned int*)&vt_lo[(dt * 16 + ln) * SA];
#pragma unroll
            for (int jj = 0; jj < 8; ++jj) {
                int off = jj + ((jj >= 4) ? 4 : 0) + 4 * half;
                vb.u[jj] = ph[off]; vl.u[jj] = pl[off];
            }
            v8f &oo = (dt == 0) ? o0 : (dt == 1) ? o1 : (dt == 2) ? o2 : o3;
            oo = wmma_bf(pa,  vb, oo);
            oo = wmma_bf(pl2, vb, oo);
            oo = wmma_bf(pa,  vl, oo);
        }
        __syncthreads();
    }

    // normalize, write merged-head output pre-split bf16 [B*S][NX]
    const int b  = bh / NHEAD;
    const int hh = bh - b * NHEAD;
#pragma unroll
    for (int v = 0; v < 8; ++v) {
        float inv = 1.0f / lrun[v];
        int s = q0 + w * 16 + v + 8 * half;
        size_t rowo = ((size_t)(b * SEQ + s)) * NX + hh * DH;
        float ov[4] = { o0[v] * inv, o1[v] * inv, o2[v] * inv, o3[v] * inv };
#pragma unroll
        for (int dt = 0; dt < 4; ++dt) {
            ushort_t h, l; splitbf(ov[dt], h, l);
            ahi[rowo + dt * 16 + ln] = h;
            alo[rowo + dt * 16 + ln] = l;
        }
    }
}

// ---------------------------------------------------------------------------
extern "C" void kernel_launch(void* const* d_in, const int* in_sizes, int n_in,
                              void* d_out, int out_size, void* d_ws, size_t ws_size,
                              hipStream_t stream) {
    (void)in_sizes; (void)n_in; (void)out_size; (void)ws_size;
    const float* x      = (const float*)d_in[0];
    const float* w_attn = (const float*)d_in[1];
    const float* b_attn = (const float*)d_in[2];
    const float* lora_A = (const float*)d_in[3];
    const float* lora_B = (const float*)d_in[4];
    const float* w_proj = (const float*)d_in[5];
    const float* b_proj = (const float*)d_in[6];
    float* out = (float*)d_out;

    const size_t MK = (size_t)MTOT * NX;      // 3,145,728

    char* p = (char*)d_ws;
    ushort_t* x_hi  = (ushort_t*)p; p += MK * 2;
    ushort_t* x_lo  = (ushort_t*)p; p += MK * 2;
    ushort_t* we_hi = (ushort_t*)p; p += (size_t)N3 * NX * 2;
    ushort_t* we_lo = (ushort_t*)p; p += (size_t)N3 * NX * 2;
    ushort_t* wp_hi = (ushort_t*)p; p += (size_t)NX * NX * 2;
    ushort_t* wp_lo = (ushort_t*)p; p += (size_t)NX * NX * 2;
    float*    qws   = (float*)p;    p += MK * 4;
    ushort_t* k_hi  = (ushort_t*)p; p += MK * 2;
    ushort_t* k_lo  = (ushort_t*)p; p += MK * 2;
    ushort_t* vt_hi = (ushort_t*)p; p += MK * 2;
    ushort_t* vt_lo = (ushort_t*)p; p += MK * 2;
    ushort_t* a_hi  = (ushort_t*)p; p += MK * 2;
    ushort_t* a_lo  = (ushort_t*)p; p += MK * 2;

    float* a_out = out;                       // [B,S,NX]
    float* k_out = out + MK;                  // present[0] = k [B,H,S,dh]
    float* v_out = k_out + MK;                // present[1] = v

    prep_split<<<(int)((MK + 255) / 256), 256, 0, stream>>>(x, x_hi, x_lo, (int)MK);
    weffT_kernel<<<(N3 * NX + 255) / 256, 256, 0, stream>>>(w_attn, lora_A, lora_B,
                                                            we_hi, we_lo);
    wT_kernel<<<(NX * NX + 255) / 256, 256, 0, stream>>>(w_proj, wp_hi, wp_lo, NX, NX);

    dim3 g1(N3 / 64, MTOT / 64);
    gemm_kernel<0><<<g1, 256, 0, stream>>>(x_hi, x_lo, we_hi, we_lo, b_attn,
                                           qws, k_out, v_out, k_hi, k_lo, N3, NX);

    dim3 gv(SEQ / 64, BATCH * NHEAD);
    vtrans_kernel<<<gv, 256, 0, stream>>>(v_out, vt_hi, vt_lo);

    dim3 g2(SEQ / 128, BATCH * NHEAD);
    attn_kernel<<<g2, 256, 0, stream>>>(qws, k_hi, k_lo, vt_hi, vt_lo, a_hi, a_lo);

    dim3 g3(NX / 64, MTOT / 64);
    gemm_kernel<1><<<g3, 256, 0, stream>>>(a_hi, a_lo, wp_hi, wp_lo, b_proj,
                                           a_out, nullptr, nullptr, nullptr, nullptr,
                                           NX, NX);
}
